// GroupedQueryAttention_32676111188216
// MI455X (gfx1250) — matrix operand order repaired and checked by reading, not compiled, not run on hardware
//
#include <hip/hip_runtime.h>

// ---------------------------------------------------------------------------
// CDNA5 / gfx1250 GQA forward: all GEMM-shaped work on v_wmma_f32_16x16x32_bf16
// GEMM B-panels staged through LDS with global_load_async_to_lds_b128 (ASYNCcnt)
// ---------------------------------------------------------------------------

#define BATCH 2
#define SEQ   2048
#define DIM_  1024
#define NH_Q  16
#define NH_KV 4
#define HD_   64
#define MROWS (BATCH * SEQ)   // 4096

typedef __attribute__((ext_vector_type(16))) __bf16 v16bf;
typedef __attribute__((ext_vector_type(8)))  float  v8f;

union Frag16 {
    v16bf v;
    uint4 q[2];
    unsigned short u[16];
};

__device__ __forceinline__ unsigned short f2bf(float f) {
    unsigned int x = __float_as_uint(f);
    unsigned int r = x + 0x7FFFu + ((x >> 16) & 1u);   // round-to-nearest-even
    return (unsigned short)(r >> 16);
}

__device__ __forceinline__ v8f bwmma(const Frag16& a, const Frag16& b, v8f c) {
    return __builtin_amdgcn_wmma_f32_16x16x32_bf16(
        false, a.v, false, b.v, (short)0, c, false, false);
}

__device__ __forceinline__ float rmax16(float v) {
    v = fmaxf(v, __shfl_xor(v, 1, 32));
    v = fmaxf(v, __shfl_xor(v, 2, 32));
    v = fmaxf(v, __shfl_xor(v, 4, 32));
    v = fmaxf(v, __shfl_xor(v, 8, 32));
    return v;
}
__device__ __forceinline__ float rsum16(float v) {
    v += __shfl_xor(v, 1, 32);
    v += __shfl_xor(v, 2, 32);
    v += __shfl_xor(v, 4, 32);
    v += __shfl_xor(v, 8, 32);
    return v;
}

// ------------------------- precision conversion ----------------------------

__global__ __launch_bounds__(256)
void to_bf16_kernel(const float* __restrict__ x, unsigned short* __restrict__ y, int n) {
    int i = blockIdx.x * blockDim.x + threadIdx.x;
    if (i < n) y[i] = f2bf(x[i]);
}

// W is [N,K] row-major; produce WT = [K,N] bf16 so B-matrix rows are contiguous
__global__ __launch_bounds__(256)
void w_transpose_bf16(const float* __restrict__ W, unsigned short* __restrict__ WT,
                      int N, int K) {
    int idx = blockIdx.x * blockDim.x + threadIdx.x;
    if (idx >= N * K) return;
    int n = idx % N, k = idx / N;
    WT[(size_t)k * N + n] = f2bf(W[(size_t)n * K + k]);
}

// ------------------------------- WMMA GEMM ---------------------------------
// out = A[M,K](bf16,row) @ BT[K,N](bf16,row) + bias.  256 thr = 8 waves;
// wave tile 16(M) x 64(N) -> block tile 32 x 256.
// B panel (32K x 256N bf16 = 16KB) is streamed into LDS per k-chunk via
// async global->LDS copies, double-buffered; WMMA B-frags read from LDS.
// All four B-fragments are loaded before the WMMA burst so the 64 16-bit LDS loads
// can be claused and their latency overlapped instead of serialized per-WMMA.
// mode 0: out[r*N+c] fp32.  mode 1: out scattered to [b, c/64, t, c%64] fp32.
__global__ __launch_bounds__(256)
void gemm_bf16_wmma(const unsigned short* __restrict__ A,
                    const unsigned short* __restrict__ BT,
                    const float* __restrict__ bias,
                    float* __restrict__ out,
                    int M, int N, int K, int mode, int T, int NHo) {
    __shared__ unsigned short bpanel[2][32][256];   // 2 x 16KB double buffer

    const int tid  = threadIdx.x;
    const int lane = tid & 31;
    const int wave = tid >> 5;
    const int half = lane >> 4;
    const int ln   = lane & 15;

    const int row0 = (blockIdx.x * 2 + (wave >> 2)) * 16;
    const int colB = blockIdx.y * 256;          // block column base (global)
    const int colW = (wave & 3) * 64;           // wave column inside panel

    const char* ldsBase = (const char*)&bpanel[0][0][0];

    // cooperative async copy of one 32(K) x 256(N) bf16 panel into LDS buffer
    auto issue_panel = [&](int k0, int buf) {
        #pragma unroll
        for (int t = 0; t < 4; ++t) {
            int id  = tid + t * 256;           // 1024 x 16B transfers total
            int row = id >> 5;                 // 32 x 16B chunks per 512B row
            int sub = id & 31;
            const unsigned short* g = BT + (size_t)(k0 + row) * N + colB + sub * 8;
            unsigned ldsOff =
                (unsigned)((const char*)&bpanel[buf][row][sub * 8] - ldsBase);
            asm volatile("global_load_async_to_lds_b128 %0, %1, off"
                         :: "v"(ldsOff), "v"(g)
                         : "memory");
        }
    };

    v8f acc[4];
    acc[0] = {}; acc[1] = {}; acc[2] = {}; acc[3] = {};

    // A layout (16-bit A 16x32): lane m = ln, chunk0 K = 8*half.., chunk1 K = 16+8*half..
    const unsigned short* arow = A + (size_t)(row0 + ln) * K;

    const int nk = K / 32;
    issue_panel(0, 0);

    for (int kc = 0; kc < nk; ++kc) {
        const int buf = kc & 1;
        // my panel loads done; barrier makes whole panel visible block-wide and
        // guarantees everyone has consumed the other buffer (WAR-safe refill).
        asm volatile("s_wait_asynccnt 0x0" ::: "memory");
        __syncthreads();
        if (kc + 1 < nk) issue_panel((kc + 1) * 32, 1 - buf);

        const int k0 = kc * 32;
        Frag16 a;
        a.q[0] = *(const uint4*)(arow + k0 + 8 * half);
        a.q[1] = *(const uint4*)(arow + k0 + 16 + 8 * half);
        __builtin_prefetch(arow + k0 + 128, 0, 3);

        // B layout: lane N = ln, element s -> K = 8*half+s (s<8), 16+8*half+s-8 (s>=8).
        // Load all 4 fragments first, then burst the 4 WMMAs.
        const unsigned short* brow = &bpanel[buf][0][colW + ln];
        Frag16 b[4];
        #pragma unroll
        for (int j = 0; j < 4; ++j) {
            #pragma unroll
            for (int s = 0; s < 16; ++s) {
                int kk = (s < 8) ? (8 * half + s) : (16 + 8 * half + s - 8);
                b[j].u[s] = brow[kk * 256 + j * 16];
            }
        }
        #pragma unroll
        for (int j = 0; j < 4; ++j)
            acc[j] = bwmma(a, b[j], acc[j]);
    }

    #pragma unroll
    for (int j = 0; j < 4; ++j) {
        #pragma unroll
        for (int i = 0; i < 8; ++i) {
            int r = row0 + i + 8 * half;      // D layout: vgpr i, M = i + 8*half
            int c = colB + colW + j * 16 + ln;//           N = lane%16
            float val = acc[j][i] + bias[c];
            if (mode == 0) {
                out[(size_t)r * N + c] = val;
            } else {
                int bb = r / T, tt = r % T;
                int hh = c >> 6, dd = c & 63;
                out[(((size_t)(bb * NHo + hh)) * T + tt) * 64 + dd] = val;
            }
        }
    }
}

// ------------------------------- RoPE pack ---------------------------------
// reference: out[i] = x[i]*cos(f_i) - x[2i+1]*sin(f_i)
//            out[i+32] = x[i+32]*cos(f_i) + x[2i]*sin(f_i),  f_i = t/10000^(i/32)

__global__ __launch_bounds__(256)
void rope_q_kernel(const float* __restrict__ qf, unsigned short* __restrict__ qb) {
    int idx = blockIdx.x * blockDim.x + threadIdx.x;   // BATCH*NH_Q*SEQ*32 threads
    int i   = idx & 31;
    int bht = idx >> 5;
    int t   = bht & (SEQ - 1);
    size_t base = (size_t)bht * HD_;
    float ang = (float)t * __expf(-(float)i * (9.210340371976184f / 32.0f));
    float c = __cosf(ang), s = __sinf(ang);
    float xi  = qf[base + i];
    float xi2 = qf[base + i + 32];
    float xe  = qf[base + 2 * i];
    float xo  = qf[base + 2 * i + 1];
    qb[base + i]      = f2bf(xi  * c - xo * s);
    qb[base + i + 32] = f2bf(xi2 * c + xe * s);
}

// k: same RoPE, but emit transposed [b, hkv, d, t] so QK^T B-frags are contiguous
__global__ __launch_bounds__(256)
void rope_k_kernel(const float* __restrict__ kf, unsigned short* __restrict__ kT) {
    int idx = blockIdx.x * blockDim.x + threadIdx.x;   // BATCH*NH_KV*SEQ*32 threads
    int i   = idx & 31;
    int bht = idx >> 5;
    int t   = bht & (SEQ - 1);
    int bh  = bht >> 11;
    size_t base = (size_t)bht * HD_;
    float ang = (float)t * __expf(-(float)i * (9.210340371976184f / 32.0f));
    float c = __cosf(ang), s = __sinf(ang);
    float xi  = kf[base + i];
    float xi2 = kf[base + i + 32];
    float xe  = kf[base + 2 * i];
    float xo  = kf[base + 2 * i + 1];
    size_t ob = (size_t)bh * HD_ * SEQ;
    kT[ob + (size_t)i * SEQ + t]        = f2bf(xi  * c - xo * s);
    kT[ob + (size_t)(i + 32) * SEQ + t] = f2bf(xi2 * c + xe * s);
}

__global__ __launch_bounds__(256)
void pack_v_kernel(const float* __restrict__ vf, unsigned short* __restrict__ vb, int n) {
    int i = blockIdx.x * blockDim.x + threadIdx.x;
    if (i < n) vb[i] = f2bf(vf[i]);
}

// --------------------------- flash attention -------------------------------
// One wave per (b, h, 16-query tile). 4 waves / block.
// grid = (BATCH*NH_Q, SEQ/64), block = 128.
__global__ __launch_bounds__(128)
void attn_kernel(const unsigned short* __restrict__ qb,   // [b,h,t,d] bf16
                 const unsigned short* __restrict__ kT,   // [b,hkv,d,t] bf16
                 const unsigned short* __restrict__ vb,   // [b,hkv,t,d] bf16
                 unsigned short* __restrict__ ctxb) {     // [b,t,h*64+d] bf16
    __shared__ unsigned short pshm[4][16][32];            // per-wave P transpose tile

    const int lane = threadIdx.x & 31;
    const int wave = threadIdx.x >> 5;
    const int half = lane >> 4;
    const int ln   = lane & 15;

    const int bh  = blockIdx.x;        // b*NH_Q + h
    const int b   = bh >> 4;
    const int h   = bh & 15;
    const int hkv = h >> 2;            // G = 4 consecutive q-heads per kv head
    const int m0  = (blockIdx.y * 4 + wave) * 16;

    const unsigned short* qbase = qb + ((size_t)bh * SEQ + m0) * HD_;
    const unsigned short* kbase = kT + (size_t)(b * NH_KV + hkv) * HD_ * SEQ;
    const unsigned short* vbase = vb + (size_t)(b * NH_KV + hkv) * SEQ * HD_;

    // Q tile as two A-fragments (d-chunks 0..31, 32..63)
    Frag16 aq0, aq1;
    {
        const unsigned short* ar = qbase + (size_t)ln * HD_;
        aq0.q[0] = *(const uint4*)(ar + 8 * half);
        aq0.q[1] = *(const uint4*)(ar + 16 + 8 * half);
        aq1.q[0] = *(const uint4*)(ar + 32 + 8 * half);
        aq1.q[1] = *(const uint4*)(ar + 48 + 8 * half);
    }

    float mrow[8], lrow[8];
    #pragma unroll
    for (int i = 0; i < 8; ++i) { mrow[i] = -3.0e38f; lrow[i] = 0.0f; }
    v8f ctx[4];
    ctx[0] = {}; ctx[1] = {}; ctx[2] = {}; ctx[3] = {};

    for (int s0 = 0; s0 < SEQ; s0 += 32) {
        // S = Q (16x64) @ K^T (64x32): load all 4 K fragments, then WMMA burst
        v8f sa = {}, sb = {};
        {
            const unsigned short* br0 = kbase + s0 + ln;
            const unsigned short* br1 = kbase + (size_t)32 * SEQ + s0 + ln;
            Frag16 b00, b01, b10, b11;
            #pragma unroll
            for (int s = 0; s < 16; ++s) {
                int kk = (s < 8) ? (8 * half + s) : (16 + 8 * half + s - 8);
                b00.u[s] = br0[(size_t)kk * SEQ];
                b01.u[s] = br0[(size_t)kk * SEQ + 16];
                b10.u[s] = br1[(size_t)kk * SEQ];
                b11.u[s] = br1[(size_t)kk * SEQ + 16];
            }
            sa = bwmma(aq0, b00, sa);
            sb = bwmma(aq0, b01, sb);
            sa = bwmma(aq1, b10, sa);
            sb = bwmma(aq1, b11, sb);
        }

        // online softmax; D layout: element i -> row i+8*half, lanes span keys
        #pragma unroll
        for (int i = 0; i < 8; ++i) {
            float x0 = sa[i] * 0.125f;                 // 1/sqrt(64)
            float x1 = sb[i] * 0.125f;
            float rm = rmax16(fmaxf(x0, x1));
            float nm = fmaxf(mrow[i], rm);
            float cf = __expf(mrow[i] - nm);
            float p0 = __expf(x0 - nm);
            float p1 = __expf(x1 - nm);
            lrow[i] = lrow[i] * cf + rsum16(p0 + p1);
            mrow[i] = nm;
            #pragma unroll
            for (int j = 0; j < 4; ++j) ctx[j][i] *= cf;
            pshm[wave][i + 8 * half][ln]      = f2bf(p0);
            pshm[wave][i + 8 * half][16 + ln] = f2bf(p1);
        }
        __asm volatile("s_wait_dscnt 0x0" ::: "memory");   // wave-private LDS RAW

        // re-read P in A-layout (lane = query row, two 8-key chunks)
        Frag16 ap;
        ap.q[0] = *(const uint4*)&pshm[wave][ln][8 * half];
        ap.q[1] = *(const uint4*)&pshm[wave][ln][16 + 8 * half];

        // ctx += P (16x32) @ V (32x64): load all 4 V fragments, then WMMA burst
        const unsigned short* vr = vbase + (size_t)s0 * HD_ + ln;
        Frag16 bv[4];
        #pragma unroll
        for (int j = 0; j < 4; ++j) {
            #pragma unroll
            for (int s = 0; s < 16; ++s) {
                int kk = (s < 8) ? (8 * half + s) : (16 + 8 * half + s - 8);
                bv[j].u[s] = vr[(size_t)kk * HD_ + j * 16];
            }
        }
        #pragma unroll
        for (int j = 0; j < 4; ++j)
            ctx[j] = bwmma(ap, bv[j], ctx[j]);
    }

    // normalize and emit bf16 context in [b, t, DIM] layout for the out-proj GEMM
    #pragma unroll
    for (int i = 0; i < 8; ++i) {
        float inv = 1.0f / lrow[i];
        int t = m0 + i + 8 * half;
        size_t ob = ((size_t)(b * SEQ + t)) * DIM_ + h * HD_;
        #pragma unroll
        for (int j = 0; j < 4; ++j)
            ctxb[ob + j * 16 + ln] = f2bf(ctx[j][i] * inv);
    }
}

// ------------------------------- launcher ----------------------------------

extern "C" void kernel_launch(void* const* d_in, const int* in_sizes, int n_in,
                              void* d_out, int out_size, void* d_ws, size_t ws_size,
                              hipStream_t stream) {
    const float* hs = (const float*)d_in[0];
    const float* Wq = (const float*)d_in[1];
    const float* bq = (const float*)d_in[2];
    const float* Wk = (const float*)d_in[3];
    const float* bk = (const float*)d_in[4];
    const float* Wv = (const float*)d_in[5];
    const float* bv = (const float*)d_in[6];
    const float* Wo = (const float*)d_in[7];
    const float* bo = (const float*)d_in[8];
    float* out = (float*)d_out;

    // ---- workspace carve-up (all 256B-aligned sizes) ----
    char* w = (char*)d_ws;
    unsigned short* hsb = (unsigned short*)w; w += (size_t)MROWS * DIM_ * 2;            // 8 MB
    unsigned short* WqT = (unsigned short*)w; w += (size_t)DIM_ * DIM_ * 2;             // 2 MB
    unsigned short* WkT = (unsigned short*)w; w += (size_t)DIM_ * 256 * 2;              // 0.5 MB
    unsigned short* WvT = (unsigned short*)w; w += (size_t)DIM_ * 256 * 2;              // 0.5 MB
    unsigned short* WoT = (unsigned short*)w; w += (size_t)DIM_ * DIM_ * 2;             // 2 MB
    float* qf = (float*)w; w += (size_t)BATCH * NH_Q  * SEQ * HD_ * 4;                  // 16 MB
    float* kf = (float*)w; w += (size_t)BATCH * NH_KV * SEQ * HD_ * 4;                  // 4 MB
    float* vf = (float*)w; w += (size_t)BATCH * NH_KV * SEQ * HD_ * 4;                  // 4 MB
    unsigned short* qbf  = (unsigned short*)w; w += (size_t)BATCH * NH_Q  * SEQ * HD_ * 2; // 8 MB
    unsigned short* kTbf = (unsigned short*)w; w += (size_t)BATCH * NH_KV * HD_ * SEQ * 2; // 2 MB
    unsigned short* vbf  = (unsigned short*)w; w += (size_t)BATCH * NH_KV * SEQ * HD_ * 2; // 2 MB
    unsigned short* ctxb = (unsigned short*)w; w += (size_t)MROWS * DIM_ * 2;           // 8 MB

    // 1) precision conversion + weight transposes
    to_bf16_kernel<<<(MROWS * DIM_) / 256, 256, 0, stream>>>(hs, hsb, MROWS * DIM_);
    w_transpose_bf16<<<(DIM_ * DIM_) / 256, 256, 0, stream>>>(Wq, WqT, DIM_, DIM_);
    w_transpose_bf16<<<(256 * DIM_) / 256, 256, 0, stream>>>(Wk, WkT, 256, DIM_);
    w_transpose_bf16<<<(256 * DIM_) / 256, 256, 0, stream>>>(Wv, WvT, 256, DIM_);
    w_transpose_bf16<<<(DIM_ * DIM_) / 256, 256, 0, stream>>>(Wo, WoT, DIM_, DIM_);

    // 2) QKV projections (WMMA + async-LDS B panels), fp32 out in [b,h,t,d]
    dim3 blk(256);
    gemm_bf16_wmma<<<dim3(MROWS / 32, DIM_ / 256), blk, 0, stream>>>(
        hsb, WqT, bq, qf, MROWS, DIM_, DIM_, 1, SEQ, NH_Q);
    gemm_bf16_wmma<<<dim3(MROWS / 32, 1), blk, 0, stream>>>(
        hsb, WkT, bk, kf, MROWS, 256, DIM_, 1, SEQ, NH_KV);
    gemm_bf16_wmma<<<dim3(MROWS / 32, 1), blk, 0, stream>>>(
        hsb, WvT, bv, vf, MROWS, 256, DIM_, 1, SEQ, NH_KV);

    // 3) RoPE + bf16 packing (k transposed for contiguous QK^T B-frags)
    rope_q_kernel<<<(BATCH * NH_Q * SEQ * 32) / 256, 256, 0, stream>>>(qf, qbf);
    rope_k_kernel<<<(BATCH * NH_KV * SEQ * 32) / 256, 256, 0, stream>>>(kf, kTbf);
    pack_v_kernel<<<(BATCH * NH_KV * SEQ * HD_) / 256, 256, 0, stream>>>(
        vf, vbf, BATCH * NH_KV * SEQ * HD_);

    // 4) flash attention (WMMA QK^T + PV, online softmax)
    attn_kernel<<<dim3(BATCH * NH_Q, SEQ / 64), dim3(128), 0, stream>>>(
        qbf, kTbf, vbf, ctxb);

    // 5) output projection (WMMA) -> fp32 d_out
    gemm_bf16_wmma<<<dim3(MROWS / 32, DIM_ / 256), blk, 0, stream>>>(
        ctxb, WoT, bo, out, MROWS, DIM_, DIM_, 0, SEQ, NH_Q);
}
